// AttnBlock_39788577030152
// MI455X (gfx1250) — matrix operand order repaired and checked by reading, not compiled, not run on hardware
//
#include <hip/hip_runtime.h>

// ---------------------------------------------------------------------------
// AttnBlock (B=2, C=64, N=16^3=4096) fused for MI455X (gfx1250, wave32, WMMA)
//   k1: QKV projection  -> q [B][N][C], k [B][N][C], v [B][C][N] (=V^T), all f16
//   k2: flash-attention  (v_wmma_f32_16x16x32_f16, online softmax) -> H [B][N][C] f32
//        * P/V fragments are contiguous 32B-per-lane loads, Q/K two 16B chunks (no packing)
//        * row-max broadcast via DPP16 butterfly (VALU, co-executes with WMMA)
//        * row-sum kept lane-partial, reduced once in the epilogue
//   k3: Wp proj + residual -> y [B][C][N] f32
//   k4: GroupNorm(32) + Swish -> out [B][C][N] f32
// ---------------------------------------------------------------------------

typedef __attribute__((ext_vector_type(16))) _Float16 v16h;
typedef __attribute__((ext_vector_type(8)))  float    v8f;

#define BATCH  2
#define CCH    64
#define NPOS   4096
#define NGROUP 32
#define EPSGN  1e-5f

#define PSTR 40   // P tile row stride in halves (32 data + 8 pad; 80B, 16B-aligned)

union FragU { v16h v; uint4 u[2]; };

// ---- DPP16 lane-exchange (stays within each 16-lane row; all lanes valid) ----
#define DPP_QUAD_XOR1   0xB1   // quad_perm [1,0,3,2]
#define DPP_QUAD_XOR2   0x4E   // quad_perm [2,3,0,1]
#define DPP_HALF_MIRROR 0x141  // row_half_mirror
#define DPP_ROW_MIRROR  0x140  // row_mirror

template<int CTRL>
__device__ __forceinline__ float dppf(float x) {
    return __builtin_bit_cast(float,
        __builtin_amdgcn_mov_dpp(__builtin_bit_cast(int, x), CTRL, 0xf, 0xf, true));
}
__device__ __forceinline__ float rowmax16(float x) {   // max over the 16-lane row
    x = fmaxf(x, dppf<DPP_QUAD_XOR1>(x));
    x = fmaxf(x, dppf<DPP_QUAD_XOR2>(x));
    x = fmaxf(x, dppf<DPP_HALF_MIRROR>(x));
    x = fmaxf(x, dppf<DPP_ROW_MIRROR>(x));
    return x;
}
__device__ __forceinline__ float rowsum16(float x) {   // sum over the 16-lane row
    x += dppf<DPP_QUAD_XOR1>(x);
    x += dppf<DPP_QUAD_XOR2>(x);
    x += dppf<DPP_HALF_MIRROR>(x);
    x += dppf<DPP_ROW_MIRROR>(x);
    return x;
}

// ---------------------------------------------------------------------------
// Kernel 1: q/k/v = W*x + b, f32 -> f16.
//   q -> [B][N][C]   (A-fragment friendly: rows = queries)
//   k -> [B][N][C]   (B-fragment friendly for S=Q*K^T: rows = keys)
//   v -> [B][C][N]   (B-fragment friendly for O=P*V: rows = channels)
// ---------------------------------------------------------------------------
__global__ __launch_bounds__(256) void qkv_kernel(
    const float* __restrict__ x,
    const float* __restrict__ Wq, const float* __restrict__ bq,
    const float* __restrict__ Wk, const float* __restrict__ bk,
    const float* __restrict__ Wv, const float* __restrict__ bv,
    _Float16* __restrict__ qt, _Float16* __restrict__ kC, _Float16* __restrict__ vt)
{
    int t = blockIdx.x * blockDim.x + threadIdx.x;   // (b, n)
    int b = t / NPOS, n = t % NPOS;
    float xv[CCH];
    const float* xb = x + (size_t)b * CCH * NPOS + n;
    #pragma unroll
    for (int c = 0; c < CCH; ++c) xv[c] = xb[(size_t)c * NPOS];   // coalesced across threads

    _Float16* q = qt + (size_t)t * CCH;
    _Float16* v = vt + (size_t)t * CCH;
    for (int o = 0; o < CCH; ++o) {
        float aq = bq[o], ak = bk[o], av = bv[o];
        const float* wq = Wq + o * CCH;
        const float* wk = Wk + o * CCH;
        const float* wv = Wv + o * CCH;
        #pragma unroll
        for (int c = 0; c < CCH; ++c) {            // W* reads are wave-uniform -> scalar path
            aq = fmaf(wq[c], xv[c], aq);
            ak = fmaf(wk[c], xv[c], ak);
            av = fmaf(wv[c], xv[c], av);
        }
        q[o] = (_Float16)aq;
        vt[((size_t)b * CCH + o) * NPOS + n] = (_Float16)av;
        kC[(size_t)t * CCH + o] = (_Float16)ak;
    }
}

// ---------------------------------------------------------------------------
// Kernel 2: fused flash attention. 4 waves/block, each wave owns a 16-row tile.
// ---------------------------------------------------------------------------
__global__ __launch_bounds__(128) void attn_kernel(
    const _Float16* __restrict__ qt, const _Float16* __restrict__ kC,
    const _Float16* __restrict__ vt, float* __restrict__ hbuf)
{
    __shared__ _Float16 pLds[4][16 * PSTR];   // per-wave P staging (C/D -> A layout)

    const int wave = threadIdx.x >> 5;
    const int lane = threadIdx.x & 31;
    const int half = lane >> 4;
    const int lm   = lane & 15;

    const int tile = blockIdx.x * 4 + wave;          // 0 .. B*256-1
    const int b    = tile / (NPOS / 16);
    const int row0 = (tile % (NPOS / 16)) * 16;

    _Float16* pT = pLds[wave];

    // ---- Q tile as two A-fragments: per lane = two contiguous 16B chunks ----
    // A 16x32: lane row m=lm; elements 0..7 = halves [8h..8h+7], 8..15 = [16+8h..23+8h]
    const _Float16* qrow = qt + ((size_t)(b * NPOS + row0 + lm)) * CCH;
    FragU aQ0, aQ1;
    aQ0.u[0] = *(const uint4*)(qrow +      8 * half);
    aQ0.u[1] = *(const uint4*)(qrow + 16 + 8 * half);
    aQ1.u[0] = *(const uint4*)(qrow + 32 +      8 * half);
    aQ1.u[1] = *(const uint4*)(qrow + 48 + 8 * half);

    float mrow[8], lrow[8];
    v8f  oAcc[4] = {v8f{}, v8f{}, v8f{}, v8f{}};
    #pragma unroll
    for (int v = 0; v < 8; ++v) { mrow[v] = -INFINITY; lrow[v] = 0.f; }

    // K rows for this lane: key lm (kr0) and 16+lm (kr1), channels contiguous
    const _Float16* kr0 = kC + ((size_t)(b * NPOS) + lm) * CCH;
    const _Float16* kr1 = kr0 + 16 * CCH;

    for (int j0 = 0; j0 < NPOS; j0 += 32) {
        FragU bk00, bk01, bk10, bk11;
        const _Float16* ka = kr0 + (size_t)j0 * CCH;
        const _Float16* kb = kr1 + (size_t)j0 * CCH;
        bk00.u[0] = *(const uint4*)(ka +      8 * half); bk00.u[1] = *(const uint4*)(ka + 16 + 8 * half);
        bk10.u[0] = *(const uint4*)(ka + 32 + 8 * half); bk10.u[1] = *(const uint4*)(ka + 48 + 8 * half);
        bk01.u[0] = *(const uint4*)(kb +      8 * half); bk01.u[1] = *(const uint4*)(kb + 16 + 8 * half);
        bk11.u[0] = *(const uint4*)(kb + 32 + 8 * half); bk11.u[1] = *(const uint4*)(kb + 48 + 8 * half);

        if (j0 + 32 < NPOS) {
            __builtin_prefetch(kr0 + j0 + 32, 0, 0);  // global_prefetch_b8
            __builtin_prefetch(kr1 + j0 + 32, 0, 0);
            __builtin_prefetch(vt + ((size_t)(b * NPOS + j0 + 32 + lane)) * CCH, 0, 0);
        }

        // ---- S = Q * K^T : two 16x16 subtiles, K-dim 64 = 2 WMMAs each ----
        v8f s0 = {}, s1 = {};
        s0 = __builtin_amdgcn_wmma_f32_16x16x32_f16(false, aQ0.v, false, bk00.v,
                                                    (short)0, s0, false, false);
        s0 = __builtin_amdgcn_wmma_f32_16x16x32_f16(false, aQ1.v, false, bk10.v,
                                                    (short)0, s0, false, false);
        s1 = __builtin_amdgcn_wmma_f32_16x16x32_f16(false, aQ0.v, false, bk01.v,
                                                    (short)0, s1, false, false);
        s1 = __builtin_amdgcn_wmma_f32_16x16x32_f16(false, aQ1.v, false, bk11.v,
                                                    (short)0, s1, false, false);

        // ---- online softmax (C/D layout: row m = v + 8*half, col = lm) ----
        float alpha[8];
        #pragma unroll
        for (int v = 0; v < 8; ++v) {
            float mx   = rowmax16(fmaxf(s0[v], s1[v]));   // DPP16 butterfly (VALU)
            float mnew = fmaxf(mrow[v], mx);
            alpha[v]   = __expf(mrow[v] - mnew);
            mrow[v]    = mnew;
            float p0 = __expf(s0[v] - mnew);
            float p1 = __expf(s1[v] - mnew);
            s0[v] = p0; s1[v] = p1;
            lrow[v] = lrow[v] * alpha[v] + (p0 + p1);     // lane-partial row sum
        }
        #pragma unroll
        for (int v = 0; v < 8; ++v) {
            oAcc[0][v] *= alpha[v]; oAcc[1][v] *= alpha[v];
            oAcc[2][v] *= alpha[v]; oAcc[3][v] *= alpha[v];
        }

        // ---- P: C/D layout -> row-major LDS -> A-fragment (2x ds_load_b128) ----
        #pragma unroll
        for (int v = 0; v < 8; ++v) {
            int m = v + 8 * half;
            pT[m * PSTR +      lm] = (_Float16)s0[v];
            pT[m * PSTR + 16 + lm] = (_Float16)s1[v];
        }
        asm volatile("s_wait_dscnt 0" ::: "memory");   // per-wave LDS visibility

        FragU aP;
        aP.u[0] = *(const uint4*)(pT + lm * PSTR + 16 * half);
        aP.u[1] = *(const uint4*)(pT + lm * PSTR + 16 * half + 8);

        // ---- O += P * V : 4 column tiles; V fragment = 32B contiguous/lane ----
        const _Float16* vrow = vt + ((size_t)(b * CCH + lm)) * NPOS + j0 + 16 * half;
        #pragma unroll
        for (int nt = 0; nt < 4; ++nt) {
            v16h bv = *(const v16h*)(vrow + (size_t)nt * 16 * NPOS);  // ch nt*16+lm, keys 16h..16h+15
            oAcc[nt] = __builtin_amdgcn_wmma_f32_16x16x32_f16(false, aP.v, false, bv,
                                                              (short)0, oAcc[nt], false, false);
        }
    }

    // ---- epilogue: complete row sums (DPP), normalize, H[b][row][c] f32 ----
    #pragma unroll
    for (int v = 0; v < 8; ++v) {
        float inv = 1.f / rowsum16(lrow[v]);
        int m = v + 8 * half;
        float* hrow = hbuf + ((size_t)(b * NPOS + row0 + m)) * CCH;
        hrow[ 0 + lm] = oAcc[0][v] * inv;
        hrow[16 + lm] = oAcc[1][v] * inv;
        hrow[32 + lm] = oAcc[2][v] * inv;
        hrow[48 + lm] = oAcc[3][v] * inv;
    }
}

// ---------------------------------------------------------------------------
// Kernel 3: y = x + Wp*H + bp   (y layout [B][C][N])
// ---------------------------------------------------------------------------
__global__ __launch_bounds__(256) void proj_kernel(
    const float* __restrict__ x, const float* __restrict__ Wp,
    const float* __restrict__ bp, const float* __restrict__ hbuf,
    float* __restrict__ y)
{
    int t = blockIdx.x * blockDim.x + threadIdx.x;   // (b, n)
    int b = t / NPOS, n = t % NPOS;
    float hv[CCH];
    const float* hr = hbuf + (size_t)t * CCH;
    #pragma unroll
    for (int c = 0; c < CCH; ++c) hv[c] = hr[c];
    for (int o = 0; o < CCH; ++o) {
        float acc = bp[o];
        const float* wp = Wp + o * CCH;
        #pragma unroll
        for (int c = 0; c < CCH; ++c) acc = fmaf(wp[c], hv[c], acc);
        size_t idx = ((size_t)b * CCH + o) * NPOS + n;
        y[idx] = x[idx] + acc;
    }
}

// ---------------------------------------------------------------------------
// Kernel 4: GroupNorm(32 groups, biased var) + Swish. 1 block per (b, group).
// ---------------------------------------------------------------------------
__global__ __launch_bounds__(256) void gn_swish_kernel(
    const float* __restrict__ y, const float* __restrict__ gamma,
    const float* __restrict__ beta, float* __restrict__ out)
{
    const int cpg = CCH / NGROUP;                  // 2 channels per group
    const int b = blockIdx.x / NGROUP;
    const int g = blockIdx.x % NGROUP;
    const int cnt = cpg * NPOS;                    // 8192
    const float* yg = y + ((size_t)b * CCH + g * cpg) * NPOS;
    float* og = out + ((size_t)b * CCH + g * cpg) * NPOS;

    float s = 0.f, ss = 0.f;
    for (int i = threadIdx.x; i < cnt; i += 256) {
        float v = yg[i]; s += v; ss = fmaf(v, v, ss);
    }
    // intra-row (16 lane) via DPP, then cross-half + cross-wave via LDS
    s  = rowsum16(s);
    ss = rowsum16(ss);
    __shared__ float rs[16], rss[16];
    int w = threadIdx.x >> 5, ln = threadIdx.x & 31;
    if ((ln & 15) == 0) { rs[2 * w + (ln >> 4)] = s; rss[2 * w + (ln >> 4)] = ss; }
    __syncthreads();
    if (threadIdx.x == 0) {
        float S = 0.f, SS = 0.f;
        #pragma unroll
        for (int i = 0; i < 16; ++i) { S += rs[i]; SS += rss[i]; }
        rs[0] = S; rss[0] = SS;
    }
    __syncthreads();
    float mean = rs[0] / (float)cnt;
    float var  = rss[0] / (float)cnt - mean * mean;   // biased, matches jnp.var
    float rstd = rsqrtf(var + EPSGN);

    for (int i = threadIdx.x; i < cnt; i += 256) {
        int c = g * cpg + i / NPOS;
        float z = (yg[i] - mean) * rstd * gamma[c] + beta[c];
        og[i] = z / (1.f + __expf(-z));               // z * sigmoid(z)
    }
}

// ---------------------------------------------------------------------------
extern "C" void kernel_launch(void* const* d_in, const int* in_sizes, int n_in,
                              void* d_out, int out_size, void* d_ws, size_t ws_size,
                              hipStream_t stream) {
    const float* x     = (const float*)d_in[0];
    const float* Wq    = (const float*)d_in[1];
    const float* bq    = (const float*)d_in[2];
    const float* Wk    = (const float*)d_in[3];
    const float* bk    = (const float*)d_in[4];
    const float* Wv    = (const float*)d_in[5];
    const float* bv    = (const float*)d_in[6];
    const float* Wp    = (const float*)d_in[7];
    const float* bp    = (const float*)d_in[8];
    const float* gamma = (const float*)d_in[9];
    const float* beta  = (const float*)d_in[10];
    float* out = (float*)d_out;

    // workspace: q,k,v (f16, 1MB each) + H (f32, 2MB) + y (f32, 2MB) = 7MB
    char* ws = (char*)d_ws;
    const size_t qkvBytes = (size_t)BATCH * NPOS * CCH * sizeof(_Float16);
    const size_t f32Bytes = (size_t)BATCH * NPOS * CCH * sizeof(float);
    _Float16* qt = (_Float16*)(ws);
    _Float16* kC = (_Float16*)(ws + qkvBytes);
    _Float16* vt = (_Float16*)(ws + 2 * qkvBytes);
    float* hbuf  = (float*)(ws + 3 * qkvBytes);
    float* ybuf  = (float*)(ws + 3 * qkvBytes + f32Bytes);

    qkv_kernel <<<BATCH * NPOS / 256, 256, 0, stream>>>(x, Wq, bq, Wk, bk, Wv, bv, qt, kC, vt);
    attn_kernel<<<BATCH * NPOS / 64,  128, 0, stream>>>(qt, kC, vt, hbuf);
    proj_kernel<<<BATCH * NPOS / 256, 256, 0, stream>>>(x, Wp, bp, hbuf, ybuf);
    gn_swish_kernel<<<BATCH * NGROUP, 256, 0, stream>>>(ybuf, gamma, beta, out);
}
